// DynamicInfiniteHeadAttention_89395449299720
// MI455X (gfx1250) — matrix operand order repaired and checked by reading, not compiled, not run on hardware
//
#include <hip/hip_runtime.h>
#include <hip/hip_bf16.h>

// ---------------------------------------------------------------------------
// DynamicInfiniteHeadAttention for MI455X (gfx1250, wave32, WMMA + TDM).
//
// All matmuls use v_wmma_f32_16x16x32_bf16 (f32 accumulate). Operands are
// pre-packed into WMMA fragments with contiguous k (lane half h holds k = 16h..16h+15) so the hot
// loops are b128 loads + WMMA. The projection kernel stages its x-tile into
// LDS with the Tensor Data Mover (tensor_load_to_lds + s_wait_tensorcnt).
// Bases/fragments (~12MB) live in the 192MB L2.
// ---------------------------------------------------------------------------

typedef __attribute__((ext_vector_type(16))) __bf16 v16bf;
typedef __attribute__((ext_vector_type(8)))  __bf16 v8bf;
typedef __attribute__((ext_vector_type(8)))  float  v8f;
typedef __attribute__((ext_vector_type(4)))  unsigned int u32x4;
typedef __attribute__((ext_vector_type(8)))  unsigned int u32x8;

#define B_   2
#define T_   4096
#define D_   1024
#define P_   16
#define H_   64
#define BT_  8192   // B_*T_

static __device__ __forceinline__ float redMax16(float v) {
#pragma unroll
  for (int m = 1; m < 16; m <<= 1) v = fmaxf(v, __shfl_xor(v, m, 16));
  return v;
}
static __device__ __forceinline__ float redSum16(float v) {
#pragma unroll
  for (int m = 1; m < 16; m <<= 1) v += __shfl_xor(v, m, 16);
  return v;
}
static __device__ __forceinline__ float redSum32(float v) {
#pragma unroll
  for (int m = 1; m < 32; m <<= 1) v += __shfl_xor(v, m, 32);
  return v;
}

static __device__ __forceinline__ v16bf cat8(v8bf lo, v8bf hi) {
  return __builtin_shufflevector(lo, hi, 0,1,2,3,4,5,6,7,8,9,10,11,12,13,14,15);
}

// ---------------------------------------------------------------------------
// Kernel 0: x (f32) -> xbf (bf16), row-major [BT][D].  Feeds the TDM in the
// projection kernel and halves proj's global read traffic.
// ---------------------------------------------------------------------------
__global__ void prep_xbf_kernel(const float* __restrict__ src,
                                __bf16* __restrict__ dst) {
  const size_t idx = (size_t)blockIdx.x * 256u + threadIdx.x;  // BT*D threads
  dst[idx] = (__bf16)src[idx];
}

// ---------------------------------------------------------------------------
// Kernel 1: router.  One wave32 per token.
// routing = softmax(softmax(rmsnorm(x,w) @ k_proto^T / 32) @ v_proto)
// ---------------------------------------------------------------------------
__global__ void router_kernel(const float* __restrict__ x,
                              const float* __restrict__ qw,
                              const float* __restrict__ kproto,
                              const float* __restrict__ vproto,
                              float* __restrict__ routing) {
  const int lane = threadIdx.x & 31;
  const int wv   = threadIdx.x >> 5;
  const int tok  = blockIdx.x * 8 + wv;

  float xv[32], xw[32];
  float ssq = 0.f;
#pragma unroll
  for (int i = 0; i < 32; ++i) {
    float v = x[tok * D_ + i * 32 + lane];
    xv[i] = v;
    xw[i] = v * qw[i * 32 + lane];
    ssq += v * v;
  }
  ssq = redSum32(ssq);
  const float rstd = rsqrtf(ssq * (1.0f / D_) + 1e-6f);

  float logit[16];
#pragma unroll
  for (int p = 0; p < 16; ++p) {
    float d = 0.f;
#pragma unroll
    for (int i = 0; i < 32; ++i) d += xw[i] * kproto[p * D_ + i * 32 + lane];
    d = redSum32(d);
    logit[p] = d * rstd * (1.0f / 32.0f);   // 1/sqrt(D)
  }
  float mx = -1e30f;
#pragma unroll
  for (int p = 0; p < 16; ++p) mx = fmaxf(mx, logit[p]);
  float aw[16]; float s = 0.f;
#pragma unroll
  for (int p = 0; p < 16; ++p) { aw[p] = __expf(logit[p] - mx); s += aw[p]; }
  float inv = 1.0f / s;
#pragma unroll
  for (int p = 0; p < 16; ++p) aw[p] *= inv;
  float r2[16];
#pragma unroll
  for (int q = 0; q < 16; ++q) {
    float d = 0.f;
#pragma unroll
    for (int p = 0; p < 16; ++p) d += aw[p] * vproto[p * 16 + q];
    r2[q] = d;
  }
  mx = -1e30f;
#pragma unroll
  for (int q = 0; q < 16; ++q) mx = fmaxf(mx, r2[q]);
  s = 0.f;
#pragma unroll
  for (int q = 0; q < 16; ++q) { r2[q] = __expf(r2[q] - mx); s += r2[q]; }
  inv = 1.0f / s;
#pragma unroll
  for (int q = 0; q < 16; ++q) {
    if (lane == q) routing[tok * 16 + q] = r2[q] * inv;
  }
}

// ---------------------------------------------------------------------------
// Kernel 2: pack a basis [P][D][H] (f32) into bf16 B-fragments.
// Layout: frag f = ((p*32+kb)*4+nb); elem addr ((f*32)+lane)*16 + j.
// B frag (32x16 bf16): lane<16: K=j, N=lane ; lane>=16: K=16+j, N=lane-16.
// ---------------------------------------------------------------------------
__global__ void prep_basis_kernel(const float* __restrict__ src,
                                  __bf16* __restrict__ dst) {
  const unsigned idx = blockIdx.x * 256u + threadIdx.x;   // 2^20 total
  const int j    = idx & 15;
  const int lane = (idx >> 4) & 31;
  const int nb   = (idx >> 9) & 3;
  const int kbk  = (idx >> 11) & 31;
  const int p    = idx >> 16;
  const int K = j + ((lane >= 16) ? 16 : 0);
  const int d = kbk * 32 + K;
  const int h = nb * 16 + (lane & 15);
  dst[idx] = (__bf16)src[(p * D_ + d) * H_ + h];
}

// ---------------------------------------------------------------------------
// Kernel 3: fused routed projection.  grid (512 token tiles, 3 bases),
// block = 128 (4 waves); wave w owns the 16-column N-tile w of H=64.
// q[t,h] = sum_p routing[t,p] * (x[t,:] @ basis[p,:,h])
// The 16x1024 bf16 x-tile is DMA'd into LDS by the Tensor Data Mover.
// ---------------------------------------------------------------------------
__global__ void proj_kernel(const __bf16* __restrict__ xbf,
                            const float* __restrict__ routing,
                            const __bf16* __restrict__ bfrags,
                            float* __restrict__ qkv) {
  __shared__ __align__(16) __bf16 xt[16][D_];
  __shared__ float rt[16][16];
  const int tile  = blockIdx.x;
  const int basis = blockIdx.y;
  const int tid   = threadIdx.x;
  const int lane  = tid & 31;
  const int wv    = tid >> 5;

  // --- TDM: DMA the 16x1024 bf16 tile (32KB) global -> LDS (wave 0 only;
  // EXEC is ignored by tensor ops, so gate per-wave to avoid duplicates).
  if (wv == 0) {
    const unsigned long long ga =
        (unsigned long long)(const void*)(xbf + (size_t)tile * 16 * D_);
    const unsigned lds = (unsigned)(unsigned long long)(uintptr_t)&xt[0][0];
    u32x4 g0;
    g0[0] = 1u;                                   // count=1 (valid), 2D, no gather
    g0[1] = lds;                                  // lds_addr        [63:32]
    g0[2] = (unsigned)ga;                         // global_addr lo  [95:64]
    g0[3] = (unsigned)((ga >> 32) & 0x1FFFFFFu)   // global_addr hi  [120:96]
            | (2u << 30);                         // type=2 (image)  [127:126]
    u32x8 g1;
    g1[0] = (1u << 16);                           // data_size=1 (2 bytes/elem)
    g1[1] = ((unsigned)D_ << 16);                 // tensor_dim0 lo16 in [63:48]
    g1[2] = ((unsigned)(BT_ & 0xFFFF) << 16);     // dim0 hi=0 | tensor_dim1 lo16
    g1[3] = ((unsigned)D_ << 16) | (BT_ >> 16);   // tensor_dim1 hi | tile_dim0
    g1[4] = 16u;                                  // tile_dim1=16, tile_dim2=0
    g1[5] = (unsigned)D_;                         // tensor_dim0_stride lo32
    g1[6] = 0u;                                   // stride hi | dim1_stride lo
    g1[7] = 0u;
    asm volatile("tensor_load_to_lds %0, %1" :: "s"(g0), "s"(g1) : "memory");
  }
  // stage routing weights while the DMA flies
  for (int i = tid; i < 256; i += 128)
    rt[i >> 4][i & 15] = routing[(tile * 16 + (i >> 4)) * 16 + (i & 15)];
  if (wv == 0) __builtin_amdgcn_s_wait_tensorcnt(0);
  __syncthreads();

  const __bf16* bf = bfrags + (size_t)basis * (1u << 20);
  const int m  = lane & 15;
  const int c0 = (lane >= 16) ? 16 : 0;
  const int hf = (lane >= 16) ? 8 : 0;

  v8f out = {};
  for (int p = 0; p < 16; ++p) {
    v8f acc = {};
    const __bf16* bp = bf + (size_t)(((p * 32) * 4 + wv) * 32) * 16;
#pragma unroll 4
    for (int kb = 0; kb < 32; ++kb) {
      v8bf lo = *(const v8bf*)&xt[m][kb * 32 + c0];
      v8bf hi = *(const v8bf*)&xt[m][kb * 32 + c0 + 8];
      v16bf a = cat8(lo, hi);
      v16bf b = *(const v16bf*)(bp + (size_t)((kb * 4) * 32 + lane) * 16);
      __builtin_prefetch(bp + (size_t)(((kb + 1) * 4) * 32 + lane) * 16, 0, 1);
      acc = __builtin_amdgcn_wmma_f32_16x16x32_bf16(false, a, false, b,
                                                    (short)0, acc, false, false);
    }
#pragma unroll
    for (int r = 0; r < 8; ++r) out[r] += rt[r + hf][p] * acc[r];
  }
  float* dst = qkv + (size_t)basis * BT_ * H_;
#pragma unroll
  for (int r = 0; r < 8; ++r) {
    int row = tile * 16 + r + hf;
    dst[row * H_ + wv * 16 + m] = out[r];
  }
}

// ---------------------------------------------------------------------------
// Kernel 4: RoPE in place on [BT][64].
// ---------------------------------------------------------------------------
__global__ void rope_kernel(float* __restrict__ p,
                            const float* __restrict__ cs,
                            const float* __restrict__ sn) {
  const int idx = blockIdx.x * 256 + threadIdx.x;   // BT*32 threads
  const int tok = idx >> 5;
  const int h   = idx & 31;
  const int base = tok * H_;
  float x1 = p[base + h], x2 = p[base + h + 32];
  float c1 = cs[base + h], s1 = sn[base + h];
  float c2 = cs[base + h + 32], s2 = sn[base + h + 32];
  p[base + h]      = x1 * c1 - x2 * s1;
  p[base + h + 32] = x2 * c2 + x1 * s2;
}

// ---------------------------------------------------------------------------
// Kernel 5a/5b/5c: pack q (A-frag, scaled by 1/sqrt(H)), k^T (B-frag),
// v (B-frag) for attention.
// A frag (16x32 bf16): M=lane&15; c0=(lane>=16)?16:0; K = c0+j.
// ---------------------------------------------------------------------------
__global__ void prep_qa_kernel(const float* __restrict__ q, __bf16* __restrict__ dst) {
  const unsigned idx = blockIdx.x * 256u + threadIdx.x;   // 2^19
  const int j = idx & 15, lane = (idx >> 4) & 31;
  const int kb = (idx >> 9) & 1, qt = (idx >> 10) & 255, b = idx >> 18;
  const int M = lane & 15, c0 = (lane >= 16) ? 16 : 0;
  const int K = c0 + j;
  const int h = kb * 32 + K;
  dst[idx] = (__bf16)(q[((size_t)b * T_ + qt * 16 + M) * H_ + h] * 0.125f);
}

__global__ void prep_kb_kernel(const float* __restrict__ k, __bf16* __restrict__ dst) {
  const unsigned idx = blockIdx.x * 256u + threadIdx.x;   // 2^19
  const int j = idx & 15, lane = (idx >> 4) & 31;
  const int hb = (idx >> 9) & 1, kt = (idx >> 10) & 255, b = idx >> 18;
  const int key = kt * 16 + (lane & 15);
  const int K = j + ((lane >= 16) ? 16 : 0);
  const int h = hb * 32 + K;
  dst[idx] = (__bf16)k[((size_t)b * T_ + key) * H_ + h];
}

__global__ void prep_vb_kernel(const float* __restrict__ v, __bf16* __restrict__ dst) {
  const unsigned idx = blockIdx.x * 256u + threadIdx.x;   // 2^19
  const int j = idx & 15, lane = (idx >> 4) & 31;
  const int nb = (idx >> 9) & 3, vt = (idx >> 11) & 127, b = idx >> 18;
  const int h = nb * 16 + (lane & 15);
  const int K = j + ((lane >= 16) ? 16 : 0);
  const int key = vt * 32 + K;
  dst[idx] = (__bf16)v[((size_t)b * T_ + key) * H_ + h];
}

// ---------------------------------------------------------------------------
// Kernel 6: flash-style causal attention.  One wave32 per 16-query tile.
// grid (256 qtiles, B).  S = Q K^T via WMMA, online softmax with 16-lane
// shuffles, P re-laid-out through LDS (same-wave DS ops are in-order),
// O += P V via WMMA.
// ---------------------------------------------------------------------------
__global__ void attn_kernel(const __bf16* __restrict__ qA,
                            const __bf16* __restrict__ kB,
                            const __bf16* __restrict__ vB,
                            float* __restrict__ attnout) {
  __shared__ __align__(16) __bf16 Pt[16][32];
  const int qt = blockIdx.x, b = blockIdx.y;
  const int lane = threadIdx.x;
  const int m = lane & 15;
  const int c0 = (lane >= 16) ? 16 : 0;
  const int hf = (lane >= 16) ? 8 : 0;

  v16bf qa0 = *(const v16bf*)(qA + (size_t)(((b * 256 + qt) * 2 + 0) * 32 + lane) * 16);
  v16bf qa1 = *(const v16bf*)(qA + (size_t)(((b * 256 + qt) * 2 + 1) * 32 + lane) * 16);

  v8f acc0 = {}, acc1 = {}, acc2 = {}, acc3 = {};
  float mr[8], lr[8];
#pragma unroll
  for (int r = 0; r < 8; ++r) { mr[r] = -1e30f; lr[r] = 0.f; }

  const int nkt = (qt * 16 + 15) / 32 + 1;
  for (int kt = 0; kt < nkt; ++kt) {
    v8f s0 = {}, s1 = {};
    {
      const size_t base0 = (size_t)(((b * 256 + 2 * kt) * 2 + 0) * 32 + lane) * 16;
      const size_t base1 = (size_t)(((b * 256 + 2 * kt + 1) * 2 + 0) * 32 + lane) * 16;
      v16bf k00 = *(const v16bf*)(kB + base0);
      v16bf k01 = *(const v16bf*)(kB + base0 + 32 * 16);
      v16bf k10 = *(const v16bf*)(kB + base1);
      v16bf k11 = *(const v16bf*)(kB + base1 + 32 * 16);
      s0 = __builtin_amdgcn_wmma_f32_16x16x32_bf16(false, qa0, false, k00, (short)0, s0, false, false);
      s0 = __builtin_amdgcn_wmma_f32_16x16x32_bf16(false, qa1, false, k01, (short)0, s0, false, false);
      s1 = __builtin_amdgcn_wmma_f32_16x16x32_bf16(false, qa0, false, k10, (short)0, s1, false, false);
      s1 = __builtin_amdgcn_wmma_f32_16x16x32_bf16(false, qa1, false, k11, (short)0, s1, false, false);
    }
    const int kbase = kt * 32;
    float scale[8];
#pragma unroll
    for (int r = 0; r < 8; ++r) {
      const int qidx = qt * 16 + r + hf;
      const float v0 = (kbase + m      <= qidx) ? s0[r] : -1e30f;
      const float v1 = (kbase + 16 + m <= qidx) ? s1[r] : -1e30f;
      float rmax = redMax16(fmaxf(v0, v1));
      const float mn = fmaxf(mr[r], rmax);
      const float sc = __expf(mr[r] - mn);
      const float p0 = __expf(v0 - mn);
      const float p1 = __expf(v1 - mn);
      const float rs = redSum16(p0 + p1);
      lr[r] = lr[r] * sc + rs;
      mr[r] = mn;
      scale[r] = sc;
      Pt[r + hf][m]      = (__bf16)p0;
      Pt[r + hf][16 + m] = (__bf16)p1;
    }
#pragma unroll
    for (int r = 0; r < 8; ++r) {
      acc0[r] *= scale[r]; acc1[r] *= scale[r];
      acc2[r] *= scale[r]; acc3[r] *= scale[r];
    }
    v8bf plo = *(const v8bf*)&Pt[m][c0];
    v8bf phi = *(const v8bf*)&Pt[m][c0 + 8];
    v16bf pa = cat8(plo, phi);
    const size_t vbase = (size_t)(((b * 128 + kt) * 4) * 32 + lane) * 16;
    v16bf vb0 = *(const v16bf*)(vB + vbase);
    v16bf vb1 = *(const v16bf*)(vB + vbase + 1 * 32 * 16);
    v16bf vb2 = *(const v16bf*)(vB + vbase + 2 * 32 * 16);
    v16bf vb3 = *(const v16bf*)(vB + vbase + 3 * 32 * 16);
    acc0 = __builtin_amdgcn_wmma_f32_16x16x32_bf16(false, pa, false, vb0, (short)0, acc0, false, false);
    acc1 = __builtin_amdgcn_wmma_f32_16x16x32_bf16(false, pa, false, vb1, (short)0, acc1, false, false);
    acc2 = __builtin_amdgcn_wmma_f32_16x16x32_bf16(false, pa, false, vb2, (short)0, acc2, false, false);
    acc3 = __builtin_amdgcn_wmma_f32_16x16x32_bf16(false, pa, false, vb3, (short)0, acc3, false, false);
  }
#pragma unroll
  for (int r = 0; r < 8; ++r) {
    const float inv = 1.0f / lr[r];
    const size_t row = ((size_t)b * T_ + qt * 16 + r + hf) * H_;
    attnout[row +  0 + m] = acc0[r] * inv;
    attnout[row + 16 + m] = acc1[r] * inv;
    attnout[row + 32 + m] = acc2[r] * inv;
    attnout[row + 48 + m] = acc3[r] * inv;
  }
}

// ---------------------------------------------------------------------------
// Kernel 7: reduce o_w over the 16 identical head slices, pack as B-frags.
// ow_red[o][h] = sum_n o_w[o, n*64+h];  out = attn @ ow_red^T.
// ---------------------------------------------------------------------------
__global__ void prep_ow_kernel(const float* __restrict__ ow, __bf16* __restrict__ dst) {
  const unsigned idx = blockIdx.x * 256u + threadIdx.x;   // 2^16
  const int j = idx & 15, lane = (idx >> 4) & 31;
  const int hb = (idx >> 9) & 1, ob = idx >> 10;
  const int o = ob * 16 + (lane & 15);
  const int K = j + ((lane >= 16) ? 16 : 0);
  const int h = hb * 32 + K;
  float s = 0.f;
#pragma unroll
  for (int n = 0; n < 16; ++n) s += ow[(size_t)o * 1024 + n * 64 + h];
  dst[idx] = (__bf16)s;
}

// ---------------------------------------------------------------------------
// Kernel 8: output GEMM [BT x 64] @ [64 x 1024].  grid (512 token tiles,
// 16 N-groups), block 128 (4 waves; wave w owns N-tile blockIdx.y*4+w).
// ---------------------------------------------------------------------------
__global__ void out_gemm_kernel(const float* __restrict__ attn,
                                const __bf16* __restrict__ owB,
                                float* __restrict__ out) {
  __shared__ __align__(16) __bf16 at[16][64];
  const int tile = blockIdx.x;
  const int tid  = threadIdx.x;
  const int lane = tid & 31;
  const int wv   = tid >> 5;
  const int ob   = blockIdx.y * 4 + wv;

#pragma unroll
  for (int i = 0; i < 8; ++i) {
    int idx = tid + i * 128;
    at[idx >> 6][idx & 63] = (__bf16)attn[(size_t)(tile * 16 + (idx >> 6)) * 64 + (idx & 63)];
  }
  __syncthreads();

  const int m = lane & 15;
  const int c0 = (lane >= 16) ? 16 : 0;
  const int hf = (lane >= 16) ? 8 : 0;

  v8f acc = {};
#pragma unroll
  for (int kb = 0; kb < 2; ++kb) {
    v8bf lo = *(const v8bf*)&at[m][kb * 32 + c0];
    v8bf hi = *(const v8bf*)&at[m][kb * 32 + c0 + 8];
    v16bf a = cat8(lo, hi);
    v16bf b = *(const v16bf*)(owB + (size_t)((ob * 2 + kb) * 32 + lane) * 16);
    acc = __builtin_amdgcn_wmma_f32_16x16x32_bf16(false, a, false, b, (short)0, acc, false, false);
  }
#pragma unroll
  for (int r = 0; r < 8; ++r) {
    int row = tile * 16 + r + hf;
    out[(size_t)row * 1024 + ob * 16 + m] = acc[r];
  }
}

// ---------------------------------------------------------------------------
// Host-side launcher.
// ---------------------------------------------------------------------------
extern "C" void kernel_launch(void* const* d_in, const int* in_sizes, int n_in,
                              void* d_out, int out_size, void* d_ws, size_t ws_size,
                              hipStream_t stream) {
  const float* x      = (const float*)d_in[0];
  const float* cs     = (const float*)d_in[1];
  const float* sn     = (const float*)d_in[2];
  const float* qnw    = (const float*)d_in[3];
  const float* kproto = (const float*)d_in[4];
  const float* vproto = (const float*)d_in[5];
  const float* qbasis = (const float*)d_in[6];
  const float* kbasis = (const float*)d_in[7];
  const float* vbasis = (const float*)d_in[8];
  const float* ow     = (const float*)d_in[9];
  float* out = (float*)d_out;

  // workspace carve-up (~35 MB total)
  char* ws = (char*)d_ws;
  size_t off = 0;
  float*  routing   = (float*)(ws + off); off += (size_t)BT_ * 16 * 4;              // 512 KB
  float*  qkv       = (float*)(ws + off); off += (size_t)3 * BT_ * H_ * 4;          // 6 MB
  __bf16* basisFrag = (__bf16*)(ws + off); off += (size_t)3 * (1u << 20) * 2;       // 6 MB
  float*  attnout   = (float*)(ws + off); off += (size_t)BT_ * H_ * 4;              // 2 MB
  __bf16* qA        = (__bf16*)(ws + off); off += (size_t)(1u << 19) * 2;           // 1 MB
  __bf16* kB        = (__bf16*)(ws + off); off += (size_t)(1u << 19) * 2;           // 1 MB
  __bf16* vB        = (__bf16*)(ws + off); off += (size_t)(1u << 19) * 2;           // 1 MB
  __bf16* owB       = (__bf16*)(ws + off); off += (size_t)(1u << 16) * 2;           // 128 KB
  __bf16* xbf       = (__bf16*)(ws + off); off += (size_t)BT_ * D_ * 2;             // 16 MB

  float* qarr = qkv;
  float* karr = qkv + (size_t)BT_ * H_;

  // 0. x -> bf16 (row-major) for the TDM-staged projection GEMM
  prep_xbf_kernel<<<(BT_ * D_) / 256, 256, 0, stream>>>(x, xbf);

  // 1. routing
  router_kernel<<<BT_ / 8, 256, 0, stream>>>(x, qnw, kproto, vproto, routing);

  // 2. pack bases into WMMA B-fragments (bf16)
  prep_basis_kernel<<<4096, 256, 0, stream>>>(qbasis, basisFrag + 0 * (1u << 20));
  prep_basis_kernel<<<4096, 256, 0, stream>>>(kbasis, basisFrag + 1 * (1u << 20));
  prep_basis_kernel<<<4096, 256, 0, stream>>>(vbasis, basisFrag + 2 * (1u << 20));

  // 3. routed projections -> q,k,v [BT][64]
  proj_kernel<<<dim3(BT_ / 16, 3), 128, 0, stream>>>(xbf, routing, basisFrag, qkv);

  // 4. RoPE on q and k
  rope_kernel<<<(BT_ * 32) / 256, 256, 0, stream>>>(qarr, cs, sn);
  rope_kernel<<<(BT_ * 32) / 256, 256, 0, stream>>>(karr, cs, sn);

  // 5. pack attention operands into fragments
  prep_qa_kernel<<<2048, 256, 0, stream>>>(qarr, qA);
  prep_kb_kernel<<<2048, 256, 0, stream>>>(karr, kB);
  prep_vb_kernel<<<2048, 256, 0, stream>>>(qkv + (size_t)2 * BT_ * H_, vB);

  // 6. flash attention
  attn_kernel<<<dim3(T_ / 16, B_), 32, 0, stream>>>(qA, kB, vB, attnout);

  // 7. reduced o_proj weight fragments
  prep_ow_kernel<<<256, 256, 0, stream>>>(ow, owB);

  // 8. output GEMM -> d_out [BT][1024]
  out_gemm_kernel<<<dim3(BT_ / 16, 16), 128, 0, stream>>>(attnout, owB, out);

  (void)in_sizes; (void)n_in; (void)out_size; (void)ws_size;
}
